// MySelfAttention_12111807775016
// MI455X (gfx1250) — compile-verified, matrix operand order checked by reading, not run on hardware
//
#include <hip/hip_runtime.h>
#include <hip/hip_bf16.h>

typedef __attribute__((ext_vector_type(16))) __bf16 v16bf;
typedef __attribute__((ext_vector_type(2)))  __bf16 v2bf;
typedef __attribute__((ext_vector_type(8)))  float  v8f;
typedef __attribute__((ext_vector_type(4)))  unsigned int v4u;
typedef __attribute__((ext_vector_type(8)))  int   v8i;
typedef __attribute__((ext_vector_type(4)))  int   v4i;

#define N_HEAD  16
#define T_SEQ   2048
#define C_DIM   1024
#define D_HEAD  64
#define B_BATCH 4

// Tensor Data Mover availability (device pass only; host pass lacks builtin).
#if defined(__has_builtin)
#if __has_builtin(__builtin_amdgcn_tensor_load_to_lds) && \
    __has_builtin(__builtin_amdgcn_s_wait_tensorcnt)
#define USE_TDM 1
#endif
#if __has_builtin(__builtin_amdgcn_cvt_pk_bf16_f32)
#define HAVE_CVT_PK_BF16 1
#endif
#endif
#ifndef USE_TDM
#define USE_TDM 0
#endif
#ifndef HAVE_CVT_PK_BF16
#define HAVE_CVT_PK_BF16 0
#endif

// Native bf16 convert (backend emits v_cvt_*bf16_f32, RNE).
__device__ __forceinline__ unsigned short f2bf(float f) {
  return __builtin_bit_cast(unsigned short, (__bf16)f);
}
// Paired pack: one v_cvt_pk_bf16_f32 for two floats.
__device__ __forceinline__ unsigned int pack2bf(float lo, float hi) {
#if HAVE_CVT_PK_BF16
  return __builtin_bit_cast(unsigned int,
                            __builtin_amdgcn_cvt_pk_bf16_f32(lo, hi));
#else
  v2bf p;
  p[0] = (__bf16)lo;
  p[1] = (__bf16)hi;
  return __builtin_bit_cast(unsigned int, p);
#endif
}

union Frag { v16bf v; unsigned int u[8]; };

// Fragment loader for v_wmma_f32_16x16x32_bf16 operands (ISA 7.12.2 16-bit A
// layout; also serves B when stored N-major-over-K). K pairs are contiguous so
// the 8 dword loads merge into ds_load_b128 / global_load_b128 pairs.
__device__ __forceinline__ void load_frag(Frag& f, const unsigned short* base,
                                          int row0, int ld, int k0, int lane) {
  const int m  = row0 + (lane & 15);
  const int kb = k0 + ((lane >> 4) << 3);
  const unsigned short* p = base + (size_t)m * ld + kb;
#pragma unroll
  for (int v = 0; v < 8; ++v) {
    int k = ((v & 3) << 1) + ((v >> 2) << 4);
    f.u[v] = *reinterpret_cast<const unsigned int*>(p + k);
  }
}

__device__ __forceinline__ v8f vzero() {
  v8f z = {0.f, 0.f, 0.f, 0.f, 0.f, 0.f, 0.f, 0.f};
  return z;
}

// ---------------------------------------------------------------------------
// Kernel 1: qkv = x @ W_attn + b_attn -> bf16 Q/K/V in [B,H,T,D]
// Pipelined global loads; packed-dword LDS staging; coalesced epilogue.
// ---------------------------------------------------------------------------
__global__ __launch_bounds__(256)
void qkv_gemm_kernel(const float* __restrict__ x, const float* __restrict__ Wa,
                     const float* __restrict__ ba,
                     unsigned short* __restrict__ qkv) {
  __shared__ __align__(16) unsigned short As[128 * 32];   // [m][k] bf16
  __shared__ __align__(16) unsigned short Bs[128 * 32];   // [n][k] bf16 (transposed)
  __shared__ __align__(16) unsigned short Cs[128 * 128];  // epilogue staging
  const int tid = threadIdx.x, lane = tid & 31, w = tid >> 5;
  const int wm = w >> 1, wn = w & 1;        // 4x2 wave grid -> 32x64 per wave
  const int n0 = blockIdx.x * 128;
  const int m0 = blockIdx.y * 128;

  v8f acc[2][4];
#pragma unroll
  for (int i = 0; i < 2; ++i)
#pragma unroll
    for (int j = 0; j < 4; ++j) acc[i][j] = vzero();

  // Per-thread global pointers, bumped by a constant stride each K-slab.
  const float* pa[4];
  const float* pb[2];
#pragma unroll
  for (int i = 0; i < 4; ++i) {
    int linear = tid + i * 256;                       // 1024 float4 slots (A)
    int row = linear >> 3, kk = (linear & 7) << 2;
    pa[i] = &x[(size_t)(m0 + row) * C_DIM + kk];
  }
#pragma unroll
  for (int i = 0; i < 2; ++i) {
    int linear = tid + i * 256;                       // 512 k-pair tasks (B)
    int kr = (linear >> 5) << 1;
    int nn = (linear & 31) << 2;
    pb[i] = &Wa[(size_t)kr * (3 * C_DIM) + n0 + nn];
  }

  float4 ra[4], rb0[2], rb1[2];
  auto gload = [&]() {
#pragma unroll
    for (int i = 0; i < 4; ++i) {
      ra[i] = *reinterpret_cast<const float4*>(pa[i]);
      pa[i] += 32;
    }
#pragma unroll
    for (int i = 0; i < 2; ++i) {
      rb0[i] = *reinterpret_cast<const float4*>(pb[i]);
      rb1[i] = *reinterpret_cast<const float4*>(pb[i] + 3 * C_DIM);
      pb[i] += 32 * (3 * C_DIM);
    }
  };
  auto sstore = [&]() {
#pragma unroll
    for (int i = 0; i < 4; ++i) {
      int linear = tid + i * 256;
      int row = linear >> 3, kk = (linear & 7) << 2;
      *reinterpret_cast<uint2*>(&As[row * 32 + kk]) =
          make_uint2(pack2bf(ra[i].x, ra[i].y), pack2bf(ra[i].z, ra[i].w));
    }
#pragma unroll
    for (int i = 0; i < 2; ++i) {
      int linear = tid + i * 256;
      int kr = (linear >> 5) << 1;
      int nn = (linear & 31) << 2;
      *reinterpret_cast<unsigned int*>(&Bs[(nn + 0) * 32 + kr]) = pack2bf(rb0[i].x, rb1[i].x);
      *reinterpret_cast<unsigned int*>(&Bs[(nn + 1) * 32 + kr]) = pack2bf(rb0[i].y, rb1[i].y);
      *reinterpret_cast<unsigned int*>(&Bs[(nn + 2) * 32 + kr]) = pack2bf(rb0[i].z, rb1[i].z);
      *reinterpret_cast<unsigned int*>(&Bs[(nn + 3) * 32 + kr]) = pack2bf(rb0[i].w, rb1[i].w);
    }
  };

  gload();
  for (int kt = 0; kt < C_DIM / 32; ++kt) {
    sstore();
    __syncthreads();
    if (kt + 1 < C_DIM / 32) {
      gload();                              // overlaps WMMA phase
      if (kt + 2 < C_DIM / 32) {            // L2 prefetch -> global_prefetch_b8
        __builtin_prefetch(pa[0], 0, 2);
        __builtin_prefetch(pb[0], 0, 2);
      }
    }
    Frag fa[2], fb[4];
#pragma unroll
    for (int am = 0; am < 2; ++am) load_frag(fa[am], As, wm * 32 + am * 16, 32, 0, lane);
#pragma unroll
    for (int bn = 0; bn < 4; ++bn) load_frag(fb[bn], Bs, wn * 64 + bn * 16, 32, 0, lane);
#pragma unroll
    for (int am = 0; am < 2; ++am)
#pragma unroll
      for (int bn = 0; bn < 4; ++bn)
        acc[am][bn] = __builtin_amdgcn_wmma_f32_16x16x32_bf16(
            false, fa[am].v, false, fb[bn].v, (short)0, acc[am][bn], false, false);
    __syncthreads();
  }

  // Epilogue: +bias, stage bf16 C tile in LDS, coalesced b128 scatter.
  const int rr = (lane >> 4) << 3, cc = lane & 15;
  float biasv[4];
#pragma unroll
  for (int bn = 0; bn < 4; ++bn) biasv[bn] = ba[n0 + wn * 64 + bn * 16 + cc];
#pragma unroll
  for (int am = 0; am < 2; ++am)
#pragma unroll
    for (int bn = 0; bn < 4; ++bn)
#pragma unroll
      for (int e = 0; e < 8; ++e)
        Cs[(wm * 32 + am * 16 + e + rr) * 128 + wn * 64 + bn * 16 + cc] =
            f2bf(acc[am][bn][e] + biasv[bn]);
  __syncthreads();
  const size_t headSz = (size_t)B_BATCH * N_HEAD * T_SEQ * D_HEAD;
#pragma unroll
  for (int i = 0; i < 8; ++i) {
    int slot = tid + i * 256;                 // 2048 uint4 slots
    int row = slot >> 4;
    int col8 = (slot & 15) << 3;              // 8-half chunks stay in one head
    uint4 val = *reinterpret_cast<const uint4*>(&Cs[row * 128 + col8]);
    int n = n0 + col8;
    int which = n >> 10, c = n & 1023;
    int h = c >> 6, d = c & 63;
    int m = m0 + row, t = m & (T_SEQ - 1), b = m >> 11;
    size_t dst = (size_t)which * headSz +
                 ((((size_t)b * N_HEAD + h) * T_SEQ + t) * D_HEAD + d);
    *reinterpret_cast<uint4*>(&qkv[dst]) = val;
  }
}

// ---------------------------------------------------------------------------
// Kernel 2: flash attention, 128-query tile per block (8 waves x 16 rows).
// K chunk staged via Tensor Data Mover (when available); V staged transposed
// with packed key-pair dwords; all staging shared by 8 waves.
// ---------------------------------------------------------------------------
__global__ __launch_bounds__(256)
void flash_attn_kernel(const unsigned short* __restrict__ qkv,
                       const int* __restrict__ amask,
                       unsigned short* __restrict__ y) {
  __shared__ __align__(16) unsigned short VT[64 * 64];     // V chunk [d][key]
  __shared__ __align__(16) unsigned short Pb[8][16 * 64];  // per-wave P; O staging
#if USE_TDM
  __shared__ __align__(16) unsigned short KT[64 * 64];     // K chunk [key][d]
#endif
  const int tid = threadIdx.x, lane = tid & 31, w = tid >> 5;
  const int bx = blockIdx.x;
  const int qt = bx & 15;                    // T/128 = 16 tiles
  const int h  = (bx >> 4) & 15;
  const int b  = bx >> 8;
  const size_t headSz  = (size_t)B_BATCH * N_HEAD * T_SEQ * D_HEAD;
  const size_t headOff = (((size_t)b * N_HEAD + h) * T_SEQ) * D_HEAD;
  const unsigned short* Q = qkv + headOff;
  const unsigned short* K = qkv + headSz + headOff;
  const unsigned short* V = qkv + 2 * headSz + headOff;
  const int q0 = qt * 128 + w * 16;

  Frag fq0, fq1;
  load_frag(fq0, Q, q0, D_HEAD, 0, lane);
  load_frag(fq1, Q, q0, D_HEAD, 32, lane);

  float rmax[8], rsum[8];
  v8f vo[4];
#pragma unroll
  for (int e = 0; e < 8; ++e) { rmax[e] = -3.0e38f; rsum[e] = 0.f; }
#pragma unroll
  for (int j = 0; j < 4; ++j) vo[j] = vzero();

  for (int kc = 0; kc < T_SEQ / 64; ++kc) {
    const int kb = kc * 64;

#if USE_TDM
    // Issue TDM for the K chunk (wave-uniform branch: TDM ignores EXEC).
    if (w == 0) {
      unsigned long long ga =
          (unsigned long long)(const void*)(K + (size_t)kb * D_HEAD);
      unsigned lds_off = (unsigned)(unsigned long long)(const void*)&KT[0];
      v4u g0 = {0u, 0u, 0u, 0u};
      v8i g1 = {0, 0, 0, 0, 0, 0, 0, 0};
      v4i g2 = {0, 0, 0, 0};
      v4i g3 = {0, 0, 0, 0};
      g0[0] = 1u;                                            // count=1
      g0[1] = lds_off;                                       // lds_addr
      g0[2] = (unsigned)(ga & 0xFFFFFFFFull);                // global_addr lo
      g0[3] = (unsigned)((ga >> 32) & 0x1FFFFFFull) | (2u << 30);  // hi | type=2
      g1[0] = (int)(1u << 16);                 // data_size = 2 bytes
      g1[1] = (int)(64u << 16);                // tensor_dim0 = 64 (D)
      g1[2] = (int)((unsigned)T_SEQ << 16);    // tensor_dim1 = 2048
      g1[3] = (int)(64u << 16);                // tile_dim0 = 64
      g1[4] = 64;                              // tile_dim1 = 64 keys
      g1[5] = D_HEAD;                          // tensor_dim0_stride = 64
#if __clang_major__ >= 23
      v8i g4 = {0, 0, 0, 0, 0, 0, 0, 0};
      __builtin_amdgcn_tensor_load_to_lds(g0, g1, g2, g3, g4, 0);
#else
      __builtin_amdgcn_tensor_load_to_lds(g0, g1, g2, g3, 0);
#endif
    }
#endif

    // Stage V chunk transposed: one (key-pair, d8) task per thread,
    // packed key-pair dwords -> 8 ds_store_b32 each.
    {
      int key = (tid >> 3) << 1;
      int d8 = (tid & 7) << 3;
      const unsigned short* vp = &V[(size_t)(kb + key) * D_HEAD + d8];
      uint4 v0 = *reinterpret_cast<const uint4*>(vp);
      uint4 v1 = *reinterpret_cast<const uint4*>(vp + D_HEAD);
      const unsigned short* h0 = reinterpret_cast<const unsigned short*>(&v0);
      const unsigned short* h1 = reinterpret_cast<const unsigned short*>(&v1);
#pragma unroll
      for (int q = 0; q < 8; ++q) {
        unsigned int dw = (unsigned int)h0[q] | ((unsigned int)h1[q] << 16);
        *reinterpret_cast<unsigned int*>(&VT[(d8 + q) * 64 + key]) = dw;
      }
    }
#if USE_TDM
    if (w == 0) __builtin_amdgcn_s_wait_tensorcnt(0);
#endif
    __syncthreads();

    // S = (Q K^T) / sqrt(D) + mask
    v8f s[4];
#pragma unroll
    for (int n = 0; n < 4; ++n) {
      Frag fk0, fk1;
#if USE_TDM
      load_frag(fk0, KT, n * 16, D_HEAD, 0, lane);
      load_frag(fk1, KT, n * 16, D_HEAD, 32, lane);
#else
      load_frag(fk0, K, kb + n * 16, D_HEAD, 0, lane);
      load_frag(fk1, K, kb + n * 16, D_HEAD, 32, lane);
#endif
      v8f sn = vzero();
      sn = __builtin_amdgcn_wmma_f32_16x16x32_bf16(false, fq0.v, false, fk0.v,
                                                   (short)0, sn, false, false);
      sn = __builtin_amdgcn_wmma_f32_16x16x32_bf16(false, fq1.v, false, fk1.v,
                                                   (short)0, sn, false, false);
      int mv = amask[b * T_SEQ + kb + n * 16 + (lane & 15)];
      float bias = mv ? 0.0f : -3.0e38f;
#pragma unroll
      for (int e = 0; e < 8; ++e) sn[e] = sn[e] * 0.125f + bias;  // 1/sqrt(64)
      s[n] = sn;
    }

    // Online softmax: rows live in (e, lane-half); reduce over 16-lane group
    float alpha[8];
#pragma unroll
    for (int e = 0; e < 8; ++e) {
      float me = fmaxf(fmaxf(s[0][e], s[1][e]), fmaxf(s[2][e], s[3][e]));
#pragma unroll
      for (int off = 1; off < 16; off <<= 1) me = fmaxf(me, __shfl_xor(me, off, 32));
      float mnew = fmaxf(rmax[e], me);
      alpha[e] = __expf(rmax[e] - mnew);
      rmax[e] = mnew;
      float psum = 0.f;
#pragma unroll
      for (int n = 0; n < 4; ++n) {
        float p = __expf(s[n][e] - mnew);
        s[n][e] = p;
        psum += p;
      }
#pragma unroll
      for (int off = 1; off < 16; off <<= 1) psum += __shfl_xor(psum, off, 32);
      rsum[e] = rsum[e] * alpha[e] + psum;
    }

    // Write P (bf16) to this wave's LDS tile; rescale running output
    unsigned short* Pw = Pb[w];
    const int rr = (lane >> 4) << 3, cc = lane & 15;
#pragma unroll
    for (int n = 0; n < 4; ++n)
#pragma unroll
      for (int e = 0; e < 8; ++e)
        Pw[(e + rr) * 64 + n * 16 + cc] = f2bf(s[n][e]);
#pragma unroll
    for (int j = 0; j < 4; ++j)
#pragma unroll
      for (int e = 0; e < 8; ++e) vo[j][e] *= alpha[e];
    __syncthreads();

    // O += P @ V
    Frag fp0, fp1;
    load_frag(fp0, Pw, 0, 64, 0, lane);
    load_frag(fp1, Pw, 0, 64, 32, lane);
#pragma unroll
    for (int j = 0; j < 4; ++j) {
      Frag fv0, fv1;
      load_frag(fv0, VT, j * 16, 64, 0, lane);
      load_frag(fv1, VT, j * 16, 64, 32, lane);
      vo[j] = __builtin_amdgcn_wmma_f32_16x16x32_bf16(false, fp0.v, false, fv0.v,
                                                      (short)0, vo[j], false, false);
      vo[j] = __builtin_amdgcn_wmma_f32_16x16x32_bf16(false, fp1.v, false, fv1.v,
                                                      (short)0, vo[j], false, false);
    }
    __syncthreads();
  }

  // Epilogue: normalize, stage O (128x64 bf16) in the P buffer, b128 stores.
  unsigned short* Ob = &Pb[0][0];
  const int rr = (lane >> 4) << 3, cc = lane & 15;
#pragma unroll
  for (int e = 0; e < 8; ++e) {
    float inv = 1.0f / rsum[e];
    int rowl = w * 16 + e + rr;
#pragma unroll
    for (int j = 0; j < 4; ++j)
      Ob[rowl * 64 + j * 16 + cc] = f2bf(vo[j][e] * inv);
  }
  __syncthreads();
#pragma unroll
  for (int i = 0; i < 4; ++i) {
    int slot = tid + i * 256;                 // 1024 uint4 slots
    int row = slot >> 3;
    int col8 = (slot & 7) << 3;
    uint4 val = *reinterpret_cast<const uint4*>(&Ob[row * 64 + col8]);
    *reinterpret_cast<uint4*>(
        &y[((size_t)b * T_SEQ + qt * 128 + row) * C_DIM + h * D_HEAD + col8]) = val;
  }
}

// ---------------------------------------------------------------------------
// Kernel 3: out = y @ W_proj + b_proj (fp32 out), pipelined; LDS epilogue.
// ---------------------------------------------------------------------------
__global__ __launch_bounds__(256)
void proj_gemm_kernel(const unsigned short* __restrict__ y,
                      const float* __restrict__ Wp, const float* __restrict__ bp,
                      float* __restrict__ out) {
  __shared__ __align__(16) unsigned short As[128 * 32];
  __shared__ __align__(16) unsigned short Bs[128 * 32];
  __shared__ __align__(16) float Cf[64 * 128];   // epilogue staging, 2 passes
  const int tid = threadIdx.x, lane = tid & 31, w = tid >> 5;
  const int wm = w >> 1, wn = w & 1;
  const int n0 = blockIdx.x * 128;
  const int m0 = blockIdx.y * 128;

  v8f acc[2][4];
#pragma unroll
  for (int i = 0; i < 2; ++i)
#pragma unroll
    for (int j = 0; j < 4; ++j) acc[i][j] = vzero();

  const unsigned short* pya[2];
  const float* pwb[2];
#pragma unroll
  for (int i = 0; i < 2; ++i) {
    int linear = tid + i * 256;                     // 512 uint4 slots (bf16 A)
    int row = linear >> 2, kk = (linear & 3) << 3;
    pya[i] = &y[(size_t)(m0 + row) * C_DIM + kk];
  }
#pragma unroll
  for (int i = 0; i < 2; ++i) {
    int linear = tid + i * 256;                     // 512 k-pair tasks (B)
    int kr = (linear >> 5) << 1;
    int nn = (linear & 31) << 2;
    pwb[i] = &Wp[(size_t)kr * C_DIM + n0 + nn];
  }

  uint4 raq[2];
  float4 rb0[2], rb1[2];
  auto gload = [&]() {
#pragma unroll
    for (int i = 0; i < 2; ++i) {
      raq[i] = *reinterpret_cast<const uint4*>(pya[i]);
      pya[i] += 32;
    }
#pragma unroll
    for (int i = 0; i < 2; ++i) {
      rb0[i] = *reinterpret_cast<const float4*>(pwb[i]);
      rb1[i] = *reinterpret_cast<const float4*>(pwb[i] + C_DIM);
      pwb[i] += 32 * C_DIM;
    }
  };
  auto sstore = [&]() {
#pragma unroll
    for (int i = 0; i < 2; ++i) {
      int linear = tid + i * 256;
      int row = linear >> 2, kk = (linear & 3) << 3;
      *reinterpret_cast<uint4*>(&As[row * 32 + kk]) = raq[i];
    }
#pragma unroll
    for (int i = 0; i < 2; ++i) {
      int linear = tid + i * 256;
      int kr = (linear >> 5) << 1;
      int nn = (linear & 31) << 2;
      *reinterpret_cast<unsigned int*>(&Bs[(nn + 0) * 32 + kr]) = pack2bf(rb0[i].x, rb1[i].x);
      *reinterpret_cast<unsigned int*>(&Bs[(nn + 1) * 32 + kr]) = pack2bf(rb0[i].y, rb1[i].y);
      *reinterpret_cast<unsigned int*>(&Bs[(nn + 2) * 32 + kr]) = pack2bf(rb0[i].z, rb1[i].z);
      *reinterpret_cast<unsigned int*>(&Bs[(nn + 3) * 32 + kr]) = pack2bf(rb0[i].w, rb1[i].w);
    }
  };

  gload();
  for (int kt = 0; kt < C_DIM / 32; ++kt) {
    sstore();
    __syncthreads();
    if (kt + 1 < C_DIM / 32) gload();
    Frag fa[2], fb[4];
#pragma unroll
    for (int am = 0; am < 2; ++am) load_frag(fa[am], As, wm * 32 + am * 16, 32, 0, lane);
#pragma unroll
    for (int bn = 0; bn < 4; ++bn) load_frag(fb[bn], Bs, wn * 64 + bn * 16, 32, 0, lane);
#pragma unroll
    for (int am = 0; am < 2; ++am)
#pragma unroll
      for (int bn = 0; bn < 4; ++bn)
        acc[am][bn] = __builtin_amdgcn_wmma_f32_16x16x32_bf16(
            false, fa[am].v, false, fb[bn].v, (short)0, acc[am][bn], false, false);
    __syncthreads();
  }

  // Epilogue: +bias, two 64-row passes through LDS, coalesced float4 stores.
  const int rr = (lane >> 4) << 3, cc = lane & 15;
  float biasv[4];
#pragma unroll
  for (int bn = 0; bn < 4; ++bn) biasv[bn] = bp[n0 + wn * 64 + bn * 16 + cc];
#pragma unroll
  for (int p = 0; p < 2; ++p) {
    if ((wm >> 1) == p) {
#pragma unroll
      for (int am = 0; am < 2; ++am)
#pragma unroll
        for (int bn = 0; bn < 4; ++bn)
#pragma unroll
          for (int e = 0; e < 8; ++e)
            Cf[((wm & 1) * 32 + am * 16 + e + rr) * 128 + wn * 64 + bn * 16 + cc] =
                acc[am][bn][e] + biasv[bn];
    }
    __syncthreads();
#pragma unroll
    for (int i = 0; i < 8; ++i) {
      int slot = tid + i * 256;               // 2048 float4 slots
      int row = slot >> 5;
      int col4 = (slot & 31) << 2;
      float4 val = *reinterpret_cast<const float4*>(&Cf[row * 128 + col4]);
      *reinterpret_cast<float4*>(
          &out[(size_t)(m0 + p * 64 + row) * C_DIM + n0 + col4]) = val;
    }
    __syncthreads();
  }
}

// ---------------------------------------------------------------------------
extern "C" void kernel_launch(void* const* d_in, const int* in_sizes, int n_in,
                              void* d_out, int out_size, void* d_ws, size_t ws_size,
                              hipStream_t stream) {
  const float* x     = (const float*)d_in[0];
  const int*   amask = (const int*)d_in[1];
  const float* Wa    = (const float*)d_in[2];
  const float* ba    = (const float*)d_in[3];
  const float* Wp    = (const float*)d_in[4];
  const float* bp    = (const float*)d_in[5];
  float* out = (float*)d_out;

  const size_t BTC = (size_t)B_BATCH * T_SEQ * C_DIM;
  unsigned short* qkv = (unsigned short*)d_ws;   // 3*BTC bf16
  unsigned short* yb  = qkv + 3 * BTC;           // BTC bf16

  qkv_gemm_kernel<<<dim3(3 * C_DIM / 128, (B_BATCH * T_SEQ) / 128), 256, 0, stream>>>(
      x, Wa, ba, qkv);
  flash_attn_kernel<<<dim3(B_BATCH * N_HEAD * (T_SEQ / 128)), 256, 0, stream>>>(
      qkv, amask, yb);
  proj_gemm_kernel<<<dim3(C_DIM / 128, (B_BATCH * T_SEQ) / 128), 256, 0, stream>>>(
      yb, Wp, bp, out);
}
